// GNN_81707457839660
// MI455X (gfx1250) — compile-verified, matrix operand order checked by reading, not run on hardware
//
#include <hip/hip_runtime.h>
#include <stdint.h>

// ---------------------------------------------------------------- constants
#define G_    64
#define P_    512
#define EPG_  4096
#define N_    (G_ * P_)    // 32768 nodes
#define E_    (G_ * EPG_)  // 262144 edges
#define H_    128

typedef __attribute__((ext_vector_type(16))) _Float16 v16h;
typedef __attribute__((ext_vector_type(8)))  _Float16 v8h;
typedef __attribute__((ext_vector_type(8)))  float    v8f;

// ------------------------------------------------------------- WMMA helpers
// Fragment load matching the CDNA5 16-bit A-matrix 16x32 layout:
// lanes 0-15 take K = k0+{0..7, 16..23}, lanes 16-31 take K = k0+{8..15, 24..31}.
// B fragments use the same pattern against transposed (N-major) f16 weights.
__device__ __forceinline__ v16h frag_ld(const _Float16* p) {
  v8h lo = *(const v8h*)(p);
  v8h hi = *(const v8h*)(p + 16);
  v16h r;
#pragma unroll
  for (int i = 0; i < 8; ++i) { r[i] = lo[i]; r[i + 8] = hi[i]; }
  return r;
}

__device__ __forceinline__ v8f wmma_f16(v16h a, v16h b, v8f c) {
  return __builtin_amdgcn_wmma_f32_16x16x32_f16(false, a, false, b, (short)0, c,
                                                false, false);
}

// Async global->LDS 16B copy (gfx1250 path, tracked by ASYNCcnt).
__device__ __forceinline__ void async_cp16(uint32_t lds_addr, const void* gp) {
  asm volatile("global_load_async_to_lds_b128 %0, %1, off" ::"v"(lds_addr),
               "v"((unsigned long long)(size_t)gp)
               : "memory");
}
__device__ __forceinline__ void async_drain() {
  asm volatile("s_wait_asynccnt 0x0" ::: "memory");
}

// Shared MLP body (16 rows): relu(Ain @ W1T^T + b1) @ W2T^T + b2 -> Out (+LN stats)
template <int KIN>
__device__ __forceinline__ void mlp_body(const _Float16* Ain, _Float16* Hid,
                                         float* Out, float* mv,
                                         const _Float16* __restrict__ W1T,
                                         const float* __restrict__ b1,
                                         const _Float16* __restrict__ W2T,
                                         const float* __restrict__ b2) {
  const int tid  = threadIdx.x;
  const int lane = tid & 31;
  const int nn   = lane & 15;
  const int ko   = (lane >> 4) << 3;
  const int mo   = ko;
  const int col0 = (tid >> 5) * 16;

  __builtin_prefetch(W1T + (size_t)(col0 + nn) * KIN, 0, 1);

  v8f acc = {};
  for (int k0 = 0; k0 < KIN; k0 += 32) {
    v16h a = frag_ld(Ain + (size_t)nn * KIN + k0 + ko);
    v16h b = frag_ld(W1T + (size_t)(col0 + nn) * KIN + k0 + ko);
    acc = wmma_f16(a, b, acc);
  }
  {
    const float bias = b1[col0 + nn];
#pragma unroll
    for (int rr = 0; rr < 8; ++rr) {
      float v = acc[rr] + bias;
      Hid[(size_t)(mo + rr) * H_ + col0 + nn] = (_Float16)(v > 0.f ? v : 0.f);
    }
  }
  __syncthreads();

  v8f acc2 = {};
  for (int k0 = 0; k0 < H_; k0 += 32) {
    v16h a = frag_ld(Hid + (size_t)nn * H_ + k0 + ko);
    v16h b = frag_ld(W2T + (size_t)(col0 + nn) * H_ + k0 + ko);
    acc2 = wmma_f16(a, b, acc2);
  }
  {
    const float bias = b2[col0 + nn];
#pragma unroll
    for (int rr = 0; rr < 8; ++rr)
      Out[(size_t)(mo + rr) * H_ + col0 + nn] = acc2[rr] + bias;
  }
  __syncthreads();

  if (tid < 16) {
    float s = 0.f, s2 = 0.f;
    for (int c = 0; c < H_; ++c) {
      float v = Out[tid * H_ + c];
      s += v; s2 += v * v;
    }
    float mean = s * (1.f / H_);
    float var  = s2 * (1.f / H_) - mean * mean;
    mv[tid * 2 + 0] = mean;
    mv[tid * 2 + 1] = rsqrtf(var + 1e-5f);
  }
  __syncthreads();
}

// --------------------------------------------------------------- small prep
__global__ void transpose_h_kernel(const float* __restrict__ W,
                                   _Float16* __restrict__ WT, int K, int Nc,
                                   int total) {
  int i = blockIdx.x * 256 + threadIdx.x;
  if (i < total) {
    int k = i / Nc, n = i - k * Nc;
    WT[(size_t)n * K + k] = (_Float16)W[i];
  }
}

__global__ void init_u_kernel(const float* __restrict__ u0,
                              _Float16* __restrict__ u) {
  int i = blockIdx.x * 256 + threadIdx.x;
  if (i < G_ * H_) u[i] = (_Float16)u0[i & (H_ - 1)];
}

__global__ void deg_kernel(const int* __restrict__ dst,
                           float* __restrict__ deg) {
  int i = blockIdx.x * 256 + threadIdx.x;
  if (i < E_) atomicAdd(&deg[dst[i]], 1.0f);
}

// ------------------------------------- input projections (f32 in -> f16 out)
template <int KIN>
__global__ void proj_relu_kernel(const float* __restrict__ IN,
                                 const _Float16* __restrict__ WT,
                                 const float* __restrict__ bias,
                                 _Float16* __restrict__ OUT) {
  __shared__ alignas(16) _Float16 Ain[16 * KIN];
  const int tid = threadIdx.x;
  const size_t r0 = (size_t)blockIdx.x * 16;
  for (int idx = tid; idx < 16 * KIN; idx += 256)
    Ain[idx] = (_Float16)IN[r0 * KIN + idx];
  __syncthreads();

  const int lane = tid & 31;
  const int nn = lane & 15, ko = (lane >> 4) << 3, mo = ko;
  const int col0 = (tid >> 5) * 16;
  v8f acc = {};
  for (int k0 = 0; k0 < KIN; k0 += 32) {
    v16h a = frag_ld(Ain + (size_t)nn * KIN + k0 + ko);
    v16h b = frag_ld(WT + (size_t)(col0 + nn) * KIN + k0 + ko);
    acc = wmma_f16(a, b, acc);
  }
  const float bv = bias[col0 + nn];
#pragma unroll
  for (int rr = 0; rr < 8; ++rr) {
    float v = acc[rr] + bv;
    OUT[(r0 + mo + rr) * H_ + col0 + nn] = (_Float16)(v > 0.f ? v : 0.f);
  }
}

// ---------------- edge MLP: 32 edges/block, async-LDS gather, dual A-tiles
__global__ void edge_mlp_kernel(const _Float16* __restrict__ xb,
                                _Float16* __restrict__ eb,
                                const _Float16* __restrict__ ub,
                                const int* __restrict__ src,
                                const int* __restrict__ dst,
                                const _Float16* __restrict__ W1T,
                                const float* __restrict__ b1,
                                const _Float16* __restrict__ W2T,
                                const float* __restrict__ b2,
                                const float* __restrict__ g,
                                const float* __restrict__ bn,
                                float* __restrict__ agg) {
  __shared__ alignas(16) _Float16 Ain[32 * 512];  // 32KB
  __shared__ alignas(16) _Float16 Hid[32 * H_];   // 8KB
  __shared__ float Out[32 * H_];                  // 16KB
  __shared__ float mv[64];
  const int tid = threadIdx.x;
  const int e0 = blockIdx.x * 32;
  {  // async-stage [x_src | x_dst | e | u_graph] -> 32 x 512 f16 (pure copy)
    const int r  = tid >> 3;         // 0..31 rows
    const int c0 = (tid & 7) * 64;   // 64-halves (128B) chunk, inside one seg
    const int edge = e0 + r;
    const int seg = c0 >> 7, off = c0 & 127;
    const _Float16* sp;
    if (seg == 0)      sp = xb + (size_t)src[edge] * H_ + off;
    else if (seg == 1) sp = xb + (size_t)dst[edge] * H_ + off;
    else if (seg == 2) sp = eb + (size_t)edge * H_ + off;
    else               sp = ub + (size_t)(edge >> 12) * H_ + off;  // EPG=4096
    const uint32_t lds = (uint32_t)(size_t)(Ain + r * 512 + c0);
#pragma unroll
    for (int i = 0; i < 8; ++i) async_cp16(lds + 16 * i, sp + 8 * i);
  }
  async_drain();
  __syncthreads();

  const int lane = tid & 31;
  const int nn = lane & 15, ko = (lane >> 4) << 3, mo = ko;
  const int col0 = (tid >> 5) * 16;

  // GEMM1: K=512, two A tiles share every B fragment.
  v8f acc0 = {}, acc1 = {};
  for (int k0 = 0; k0 < 512; k0 += 32) {
    v16h b  = frag_ld(W1T + (size_t)(col0 + nn) * 512 + k0 + ko);
    v16h a0 = frag_ld(Ain + (size_t)nn * 512 + k0 + ko);
    v16h a1 = frag_ld(Ain + (size_t)(nn + 16) * 512 + k0 + ko);
    acc0 = wmma_f16(a0, b, acc0);
    acc1 = wmma_f16(a1, b, acc1);
  }
  {
    const float bias = b1[col0 + nn];
#pragma unroll
    for (int rr = 0; rr < 8; ++rr) {
      float v0 = acc0[rr] + bias;
      float v1 = acc1[rr] + bias;
      Hid[(size_t)(mo + rr) * H_ + col0 + nn] = (_Float16)(v0 > 0.f ? v0 : 0.f);
      Hid[(size_t)(16 + mo + rr) * H_ + col0 + nn] =
          (_Float16)(v1 > 0.f ? v1 : 0.f);
    }
  }
  __syncthreads();

  // GEMM2: K=128
  v8f d0 = {}, d1 = {};
  for (int k0 = 0; k0 < H_; k0 += 32) {
    v16h b  = frag_ld(W2T + (size_t)(col0 + nn) * H_ + k0 + ko);
    v16h a0 = frag_ld(Hid + (size_t)nn * H_ + k0 + ko);
    v16h a1 = frag_ld(Hid + (size_t)(nn + 16) * H_ + k0 + ko);
    d0 = wmma_f16(a0, b, d0);
    d1 = wmma_f16(a1, b, d1);
  }
  {
    const float bias = b2[col0 + nn];
#pragma unroll
    for (int rr = 0; rr < 8; ++rr) {
      Out[(size_t)(mo + rr) * H_ + col0 + nn] = d0[rr] + bias;
      Out[(size_t)(16 + mo + rr) * H_ + col0 + nn] = d1[rr] + bias;
    }
  }
  __syncthreads();

  if (tid < 32) {
    float s = 0.f, s2 = 0.f;
    for (int c = 0; c < H_; ++c) {
      float v = Out[tid * H_ + c];
      s += v; s2 += v * v;
    }
    float mean = s * (1.f / H_);
    float var  = s2 * (1.f / H_) - mean * mean;
    mv[tid * 2 + 0] = mean;
    mv[tid * 2 + 1] = rsqrtf(var + 1e-5f);
  }
  __syncthreads();

  // LN + write e (f16) + fused segment_sum scatter (f32 atomics)
  const int row = tid >> 3;
  const int cc0 = (tid & 7) * 16;
  const int edge = e0 + row;
  const int dnode = dst[edge];
  const float mean = mv[row * 2 + 0], rstd = mv[row * 2 + 1];
#pragma unroll
  for (int i = 0; i < 16; ++i) {
    int c = cc0 + i;
    float v = (Out[row * H_ + c] - mean) * rstd * g[c] + bn[c];
    eb[(size_t)edge * H_ + c] = (_Float16)v;
    atomicAdd(&agg[(size_t)dnode * H_ + c], v);
  }
}

// ----------------------------------------------------------------- node MLP
__global__ void node_mlp_kernel(_Float16* __restrict__ xb,
                                const float* __restrict__ aggb,
                                const float* __restrict__ degb,
                                const _Float16* __restrict__ ub,
                                const _Float16* __restrict__ W1T,
                                const float* __restrict__ b1,
                                const _Float16* __restrict__ W2T,
                                const float* __restrict__ b2,
                                const float* __restrict__ g,
                                const float* __restrict__ bn) {
  __shared__ alignas(16) _Float16 Ain[16 * 384];
  __shared__ alignas(16) _Float16 Hid[16 * H_];
  __shared__ float Out[16 * H_];
  __shared__ float mv[32];
  const int tid = threadIdx.x;
  const int n0 = blockIdx.x * 16;
  {  // stage [x | agg/deg | u_graph] -> 16 x 384 f16
    const int r = tid >> 4;
    const int node = n0 + r;
    const int graph = node >> 9;  // P=512
    const float scale = 1.0f / fmaxf(degb[node], 1.0f);
    const int c0 = (tid & 15) * 24;
    for (int i = 0; i < 24; ++i) {
      int c = c0 + i, seg = c >> 7, off = c & 127;
      _Float16 v;
      if (seg == 0)      v = xb[(size_t)node * H_ + off];
      else if (seg == 1) v = (_Float16)(aggb[(size_t)node * H_ + off] * scale);
      else               v = ub[(size_t)graph * H_ + off];
      Ain[r * 384 + c] = v;
    }
  }
  __syncthreads();
  mlp_body<384>(Ain, Hid, Out, mv, W1T, b1, W2T, b2);

  const int row = tid >> 4;
  const int c0 = (tid & 15) * 8;
  const int node = n0 + row;
  const float mean = mv[row * 2 + 0], rstd = mv[row * 2 + 1];
#pragma unroll
  for (int i = 0; i < 8; ++i) {
    int c = c0 + i;
    xb[(size_t)node * H_ + c] =
        (_Float16)((Out[row * H_ + c] - mean) * rstd * g[c] + bn[c]);
  }
}

// --------------------------------------------------------------- global MLP
__global__ void global_mlp_kernel(_Float16* __restrict__ ub,
                                  const float* __restrict__ nmean,
                                  const float* __restrict__ emean,
                                  const _Float16* __restrict__ W1T,
                                  const float* __restrict__ b1,
                                  const _Float16* __restrict__ W2T,
                                  const float* __restrict__ b2,
                                  const float* __restrict__ g,
                                  const float* __restrict__ bn) {
  __shared__ alignas(16) _Float16 Ain[16 * 384];
  __shared__ alignas(16) _Float16 Hid[16 * H_];
  __shared__ float Out[16 * H_];
  __shared__ float mv[32];
  const int tid = threadIdx.x;
  const int g0 = blockIdx.x * 16;
  {
    const int r = tid >> 4;
    const int gr = g0 + r;
    const int c0 = (tid & 15) * 24;
    for (int i = 0; i < 24; ++i) {
      int c = c0 + i, seg = c >> 7, off = c & 127;
      float v;
      if (seg == 0)      v = (float)ub[(size_t)gr * H_ + off];
      else if (seg == 1) v = nmean[(size_t)gr * H_ + off];
      else               v = emean[(size_t)gr * H_ + off];
      Ain[r * 384 + c] = (_Float16)v;
    }
  }
  __syncthreads();
  mlp_body<384>(Ain, Hid, Out, mv, W1T, b1, W2T, b2);

  const int row = tid >> 4;
  const int c0 = (tid & 15) * 8;
  const int gr = g0 + row;
  const float mean = mv[row * 2 + 0], rstd = mv[row * 2 + 1];
#pragma unroll
  for (int i = 0; i < 8; ++i) {
    int c = c0 + i;
    ub[(size_t)gr * H_ + c] =
        (_Float16)((Out[row * H_ + c] - mean) * rstd * g[c] + bn[c]);
  }
}

// ------------------------------------------------------- per-graph mean pool
__global__ void seg_mean_kernel(const _Float16* __restrict__ V,
                                int rows_per_graph, float inv,
                                float* __restrict__ out) {
  __shared__ float acc2[H_];
  const int g = blockIdx.x, tid = threadIdx.x;
  const int col = tid & 127, half = tid >> 7;
  const int hr = rows_per_graph >> 1;
  const _Float16* base = V + (size_t)g * rows_per_graph * H_ +
                         (size_t)half * hr * H_ + col;
  float s = 0.f;
  for (int i = 0; i < hr; ++i) s += (float)base[(size_t)i * H_];
  if (half) acc2[col] = s;
  __syncthreads();
  if (!half) out[(size_t)g * H_ + col] = (s + acc2[col]) * inv;
}

// -------------------- scalar heads: relu(XW1+b1) . w2 + b2 (WMMA + block dot)
__global__ void head_kernel(const _Float16* __restrict__ X,
                            const _Float16* __restrict__ W1T,
                            const float* __restrict__ b1,
                            const float* __restrict__ w2,
                            const float* __restrict__ b2,
                            float* __restrict__ outv) {
  __shared__ alignas(16) _Float16 Ain[16 * H_];
  __shared__ float Hid[16 * H_];
  __shared__ float partial[16][16];
  const int tid = threadIdx.x;
  const size_t r0 = (size_t)blockIdx.x * 16;
  // async-stage 16x128 f16 rows (one b128 per thread)
  async_cp16((uint32_t)(size_t)Ain + tid * 16, X + r0 * H_ + tid * 8);
  async_drain();
  __syncthreads();

  const int lane = tid & 31;
  const int nn = lane & 15, ko = (lane >> 4) << 3, mo = ko;
  const int col0 = (tid >> 5) * 16;
  v8f acc = {};
  for (int k0 = 0; k0 < H_; k0 += 32) {
    v16h a = frag_ld(Ain + (size_t)nn * H_ + k0 + ko);
    v16h b = frag_ld(W1T + (size_t)(col0 + nn) * H_ + k0 + ko);
    acc = wmma_f16(a, b, acc);
  }
  const float bv = b1[col0 + nn];
#pragma unroll
  for (int rr = 0; rr < 8; ++rr) {
    float v = acc[rr] + bv;
    Hid[(mo + rr) * H_ + col0 + nn] = v > 0.f ? v : 0.f;
  }
  __syncthreads();

  const int row = tid >> 4, seg = tid & 15;
  float s = 0.f;
#pragma unroll
  for (int i = 0; i < 8; ++i) {
    int c = seg * 8 + i;
    s += Hid[row * H_ + c] * w2[c];
  }
  partial[row][seg] = s;
  __syncthreads();
  if (tid < 16) {
    float t = 0.f;
#pragma unroll
    for (int i = 0; i < 16; ++i) t += partial[tid][i];
    outv[r0 + tid] = t + b2[0];
  }
}

// --------------------------- masked softmax / argmax / entropy per graph
__global__ void softmax_kernel(const float* __restrict__ logits,
                               const unsigned char* __restrict__ mask,
                               float* __restrict__ out) {
  __shared__ float red[256];
  __shared__ int redi[256];
  const int g = blockIdx.x, tid = threadIdx.x;
  const float NEG = -3.0e38f;
  const bool ok0 = mask[(size_t)g * P_ + tid] != 0;
  const bool ok1 = mask[(size_t)g * P_ + tid + 256] != 0;
  const float l0 = ok0 ? logits[(size_t)g * P_ + tid] : NEG;
  const float l1 = ok1 ? logits[(size_t)g * P_ + tid + 256] : NEG;
  float m; int am;
  if (l0 >= l1) { m = l0; am = tid; } else { m = l1; am = tid + 256; }
  red[tid] = m; redi[tid] = am;
  __syncthreads();
  for (int s = 128; s > 0; s >>= 1) {
    if (tid < s) {
      float o = red[tid + s]; int oi = redi[tid + s];
      if (o > red[tid] || (o == red[tid] && oi < redi[tid])) {
        red[tid] = o; redi[tid] = oi;
      }
    }
    __syncthreads();
  }
  const float mx = red[0];
  const int amx = redi[0];
  __syncthreads();

  const float d0 = l0 - mx, d1 = l1 - mx;
  const float e0 = ok0 ? __expf(d0) : 0.f;
  const float e1 = ok1 ? __expf(d1) : 0.f;
  red[tid] = e0 + e1;
  __syncthreads();
  for (int s = 128; s > 0; s >>= 1) {
    if (tid < s) red[tid] += red[tid + s];
    __syncthreads();
  }
  const float s1 = red[0];
  __syncthreads();
  red[tid] = d0 * e0 + d1 * e1;
  __syncthreads();
  for (int s = 128; s > 0; s >>= 1) {
    if (tid < s) red[tid] += red[tid + s];
    __syncthreads();
  }
  const float s2 = red[0];
  if (tid == 0) {
    out[g]          = (float)amx;            // action
    out[G_ + g]     = -__logf(s1);           // logprob of argmax
    out[2 * G_ + g] = __logf(s1) - s2 / s1;  // entropy
  }
}

// ------------------------------------------------------------------- driver
extern "C" void kernel_launch(void* const* d_in, const int* in_sizes, int n_in,
                              void* d_out, int out_size, void* d_ws,
                              size_t ws_size, hipStream_t stream) {
  const float* x_in  = (const float*)d_in[0];
  const float* eattr = (const float*)d_in[1];
  const int* eindex  = (const int*)d_in[2];
  const unsigned char* mask = (const unsigned char*)d_in[4];
  const float* Wn  = (const float*)d_in[6];  const float* bn_ = (const float*)d_in[7];
  const float* We  = (const float*)d_in[8];  const float* be_ = (const float*)d_in[9];
  const float* u0  = (const float*)d_in[10];
  const float* eW1 = (const float*)d_in[11]; const float* eb1 = (const float*)d_in[12];
  const float* eW2 = (const float*)d_in[13]; const float* eb2 = (const float*)d_in[14];
  const float* eg  = (const float*)d_in[15]; const float* ebn = (const float*)d_in[16];
  const float* nW1 = (const float*)d_in[17]; const float* nb1 = (const float*)d_in[18];
  const float* nW2 = (const float*)d_in[19]; const float* nb2 = (const float*)d_in[20];
  const float* ng  = (const float*)d_in[21]; const float* nbn = (const float*)d_in[22];
  const float* gW1 = (const float*)d_in[23]; const float* gb1 = (const float*)d_in[24];
  const float* gW2 = (const float*)d_in[25]; const float* gb2 = (const float*)d_in[26];
  const float* gg  = (const float*)d_in[27]; const float* gbn = (const float*)d_in[28];
  const float* aW1 = (const float*)d_in[29]; const float* ab1 = (const float*)d_in[30];
  const float* aW2 = (const float*)d_in[31]; const float* ab2 = (const float*)d_in[32];
  const float* cW1 = (const float*)d_in[33]; const float* cb1 = (const float*)d_in[34];
  const float* cW2 = (const float*)d_in[35]; const float* cb2 = (const float*)d_in[36];
  (void)in_sizes; (void)n_in; (void)out_size; (void)ws_size;

  const int* srcI = eindex;
  const int* dstI = eindex + E_;

  char* ws = (char*)d_ws;
  size_t off = 0;
  auto carve = [&](size_t bytes) -> void* {
    void* p = ws + off;
    off += (bytes + 255) & ~(size_t)255;
    return p;
  };
  _Float16* xb   = (_Float16*)carve((size_t)N_ * H_ * 2);
  _Float16* ebf  = (_Float16*)carve((size_t)E_ * H_ * 2);
  _Float16* ub   = (_Float16*)carve((size_t)G_ * H_ * 2);
  float* aggb    = (float*)carve((size_t)N_ * H_ * 4);
  float* degb    = (float*)carve((size_t)N_ * 4);
  float* nmeanb  = (float*)carve((size_t)G_ * H_ * 4);
  float* emeanb  = (float*)carve((size_t)G_ * H_ * 4);
  float* logitsb = (float*)carve((size_t)N_ * 4);
  _Float16* WnT  = (_Float16*)carve((size_t)H_ * 64 * 2);
  _Float16* WeT  = (_Float16*)carve((size_t)H_ * 32 * 2);
  _Float16* eW1T = (_Float16*)carve((size_t)H_ * 512 * 2);
  _Float16* eW2T = (_Float16*)carve((size_t)H_ * H_ * 2);
  _Float16* nW1T = (_Float16*)carve((size_t)H_ * 384 * 2);
  _Float16* nW2T = (_Float16*)carve((size_t)H_ * H_ * 2);
  _Float16* gW1T = (_Float16*)carve((size_t)H_ * 384 * 2);
  _Float16* gW2T = (_Float16*)carve((size_t)H_ * H_ * 2);
  _Float16* aW1T = (_Float16*)carve((size_t)H_ * H_ * 2);
  _Float16* cW1T = (_Float16*)carve((size_t)H_ * H_ * 2);

  auto tr = [&](const float* W, _Float16* WT, int K, int Nc) {
    int total = K * Nc;
    transpose_h_kernel<<<(total + 255) / 256, 256, 0, stream>>>(W, WT, K, Nc,
                                                                total);
  };
  tr(Wn, WnT, 64, H_);    tr(We, WeT, 32, H_);
  tr(eW1, eW1T, 512, H_); tr(eW2, eW2T, H_, H_);
  tr(nW1, nW1T, 384, H_); tr(nW2, nW2T, H_, H_);
  tr(gW1, gW1T, 384, H_); tr(gW2, gW2T, H_, H_);
  tr(aW1, aW1T, H_, H_);  tr(cW1, cW1T, H_, H_);

  init_u_kernel<<<(G_ * H_ + 255) / 256, 256, 0, stream>>>(u0, ub);
  hipMemsetAsync(degb, 0, (size_t)N_ * 4, stream);
  deg_kernel<<<(E_ + 255) / 256, 256, 0, stream>>>(dstI, degb);

  proj_relu_kernel<64><<<N_ / 16, 256, 0, stream>>>(x_in, WnT, bn_, xb);
  proj_relu_kernel<32><<<E_ / 16, 256, 0, stream>>>(eattr, WeT, be_, ebf);

  for (int layer = 0; layer < 2; ++layer) {
    hipMemsetAsync(aggb, 0, (size_t)N_ * H_ * 4, stream);
    edge_mlp_kernel<<<E_ / 32, 256, 0, stream>>>(xb, ebf, ub, srcI, dstI, eW1T,
                                                 eb1, eW2T, eb2, eg, ebn, aggb);
    node_mlp_kernel<<<N_ / 16, 256, 0, stream>>>(xb, aggb, degb, ub, nW1T, nb1,
                                                 nW2T, nb2, ng, nbn);
    seg_mean_kernel<<<G_, 256, 0, stream>>>(xb, P_, 1.0f / P_, nmeanb);
    seg_mean_kernel<<<G_, 256, 0, stream>>>(ebf, EPG_, 1.0f / EPG_, emeanb);
    global_mlp_kernel<<<G_ / 16, 256, 0, stream>>>(ub, nmeanb, emeanb, gW1T,
                                                   gb1, gW2T, gb2, gg, gbn);
  }

  float* outp = (float*)d_out;
  head_kernel<<<N_ / 16, 256, 0, stream>>>(xb, aW1T, ab1, aW2, ab2, logitsb);
  softmax_kernel<<<G_, 256, 0, stream>>>(logitsb, mask, outp);
  head_kernel<<<G_ / 16, 256, 0, stream>>>(ub, cW1T, cb1, cW2, cb2,
                                           outp + 3 * G_);
}
